// HybridGNN_77756087927558
// MI455X (gfx1250) — matrix operand order repaired and checked by reading, not compiled, not run on hardware
//
#include <hip/hip_runtime.h>
#include <hip/hip_bf16.h>

// ---------------- problem constants ----------------
#define N_NODES 20000
#define N_EDGES 320000
#define NB      64
#define NLAYERS 4
#define DIN_SEQ 1280
#define DIN_STR 128
#define HID     256
#define NHEAD   8
#define HDIM    32
#define BN_EPS  1e-5f
#define SLOPE   0.2f

typedef __attribute__((ext_vector_type(16))) __bf16 v16bf;
typedef __attribute__((ext_vector_type(8)))  __bf16 v8bf;
typedef __attribute__((ext_vector_type(4)))  __bf16 v4bf;
typedef __attribute__((ext_vector_type(8)))  float  v8f;
typedef __attribute__((ext_vector_type(4)))  int    v4i;

// CDNA5 async global->LDS path (builtin existence + param types confirmed by
// earlier rounds: (v4i as1*, v4i as3*, imm int, imm int))
#if defined(__gfx1250__) && __has_builtin(__builtin_amdgcn_global_load_async_to_lds_b128) && \
    __has_builtin(__builtin_amdgcn_s_wait_asynccnt)
#define USE_ASYNC_LDS 1
typedef __attribute__((address_space(1))) v4i* gas1_v4i;
typedef __attribute__((address_space(3))) v4i* las3_v4i;
__device__ inline void async_g2l_b128(const void* gsrc, void* ldst) {
  __builtin_amdgcn_global_load_async_to_lds_b128(
      (gas1_v4i)(gsrc), (las3_v4i)(ldst), 0, 0);
}
#else
#define USE_ASYNC_LDS 0
#endif

// ---------------- small device helpers ----------------
__device__ inline __bf16 f2bf(float f) {            // RNE float->bf16
  unsigned u = __builtin_bit_cast(unsigned, f);
  unsigned r = (u + 0x7FFFu + ((u >> 16) & 1u)) >> 16;
  unsigned short s = (unsigned short)r;
  return __builtin_bit_cast(__bf16, s);
}
// order-preserving float<->uint encoding for atomicMax on floats
__device__ inline unsigned fenc(float f) {
  unsigned u = __builtin_bit_cast(unsigned, f);
  return (u & 0x80000000u) ? ~u : (u | 0x80000000u);
}
__device__ inline float fdec(unsigned u) {
  unsigned v = (u & 0x80000000u) ? (u ^ 0x80000000u) : ~u;
  return __builtin_bit_cast(float, v);
}
__device__ inline float lrelu(float x) { return x > 0.0f ? x : SLOPE * x; }

// ---------------- f32 -> bf16 bulk conversion (weights, once per call) -----
__global__ void cvt_f32_bf16(const float* __restrict__ in, __bf16* __restrict__ out,
                             long long n4) {   // n4 = n/4, n divisible by 4
  long long i = (long long)blockIdx.x * blockDim.x + threadIdx.x;
  if (i >= n4) return;
  const float4 v = *(const float4*)(in + i * 4);
  v4bf o; o[0] = f2bf(v.x); o[1] = f2bf(v.y); o[2] = f2bf(v.z); o[3] = f2bf(v.w);
  *(v4bf*)(out + i * 4) = o;
}

// ---------------- WMMA bf16 GEMM: C[M,NH] = act(A[M,K] @ W[K,NH] + bias) ----
// A: fp32 row-major (async-staged through LDS, block-converted to bf16);
// W: bf16 row-major (pre-converted);  C: fp32.  One wave per 16x16 tile,
// 8 waves per block share one A tile.  Fragment layouts per ISA 7.12.2:
//  A (16x32): lane L holds row (L&15); lanes<16 K{0..7,16..23}, lanes>=16
//             K{8..15,24..31}.   B (32x16): lane L holds col
//             (L&15), K split as A (16 strided 2-byte loads).
//  C/D: elem v -> row m0+(L>=16?8:0)+v, col n0+(L&15).
// All per-iteration addressing is strength-reduced to running pointers.
__global__ __launch_bounds__(256) void wmma_gemm_bias_act(
    const float* __restrict__ A, const __bf16* __restrict__ Wb,
    const float* __restrict__ bias, float* __restrict__ C,
    int M, int K, int NH, int act) {
  __shared__ __attribute__((aligned(16))) __bf16 a16[16 * 32];   // 1 KB bf16 tile
#if USE_ASYNC_LDS
  __shared__ __attribute__((aligned(16))) float a32[2][16 * 32]; // 2x2KB staging
#endif
  const int tid  = threadIdx.x;
  const int lane = tid & 31;
  const int wave = tid >> 5;
  const int m0 = blockIdx.x * 16;
  const int n0 = (blockIdx.y * 8 + wave) * 16;
  const bool active = (n0 < NH);           // wave-uniform: EXEC all-1 at WMMA

  const int brow  = (lane >> 4) * 8;
  const int khalf = (lane >> 4) * 8;
  const long long bstep = 32LL * NH;       // B pointer stride per K-step
  const __bf16* bp = Wb + (long long)brow * NH + n0 + (lane & 15);
  v8f acc = {};

#if USE_ASYNC_LDS
  const int sr = tid >> 3, sc = (tid & 7) * 4;          // staging lane mapping
  const float* asrc = A + (long long)(m0 + sr) * K + sc; // running async source
  float* sdst0 = &a32[0][sr * 32 + sc];
  float* sdst1 = &a32[1][sr * 32 + sc];
  if (tid < 128) {                         // prologue: async fill tile 0
    async_g2l_b128(asrc, sdst0);
    asrc += 32;                            // now points at tile 1
  }
  __builtin_amdgcn_s_wait_asynccnt(0);
  __syncthreads();
#endif

  const int nk = K / 32;
  for (int t = 0; t < nk; ++t) {
#if USE_ASYNC_LDS
    const float* cur = a32[t & 1];
    if (t + 1 < nk && tid < 128) {         // overlap: async fetch next tile
      async_g2l_b128(asrc, (t & 1) ? sdst0 : sdst1);
      asrc += 32;
    }
#pragma unroll
    for (int i = tid; i < 512; i += 256) a16[i] = f2bf(cur[i]);  // 2 cvt/thread
#else
    const float* arow0 = A + (long long)m0 * K + t * 32;
#pragma unroll
    for (int i = tid; i < 512; i += 256) {
      const int r = i >> 5, c = i & 31;
      a16[i] = f2bf(arow0[(long long)r * K + c]);
    }
#endif
    __syncthreads();

    if (active) {
      const __bf16* arow = a16 + (lane & 15) * 32;
      const v8bf lo = *(const v8bf*)(arow + khalf);          // ds_load_b128
      const v8bf hi = *(const v8bf*)(arow + khalf + 16);     // ds_load_b128
      const v16bf a = __builtin_shufflevector(lo, hi,
          0, 1, 2, 3, 4, 5, 6, 7, 8, 9, 10, 11, 12, 13, 14, 15);
      v16bf b;                                               // column gather, K order = A
#pragma unroll
      for (int s = 0; s < 16; ++s) b[s] = bp[(long long)(s < 8 ? s : s + 8) * NH];
      acc = __builtin_amdgcn_wmma_f32_16x16x32_bf16(
          false, a, false, b, (short)0, acc, false, false);
    }
    bp += bstep;
#if USE_ASYNC_LDS
    __builtin_amdgcn_s_wait_asynccnt(0);
#endif
    __syncthreads();
  }

  if (active) {
    const int crow = m0 + (lane >> 4) * 8;
    const int ccol = n0 + (lane & 15);
    const float bv = bias ? bias[ccol] : 0.0f;
    float* cp = C + (long long)crow * NH + ccol;             // running pointer
#pragma unroll
    for (int v = 0; v < 8; ++v) {
      float x = acc[v] + bv;
      if (act) x = fmaxf(x, 0.0f);
      *cp = x;
      cp += NH;
    }
  }
}

// ---------------- utility kernels ----------------
__global__ void zero_f32(float* p, long long n) {
  long long i = (long long)blockIdx.x * blockDim.x + threadIdx.x;
  long long st = (long long)gridDim.x * blockDim.x;
  for (; i < n; i += st) p[i] = 0.0f;
}

// ---------------- GCN pieces ----------------
__global__ void compute_deg(const int* __restrict__ ei, float* __restrict__ deg) {
  int e = blockIdx.x * blockDim.x + threadIdx.x;
  if (e >= N_EDGES + N_NODES) return;
  int dst = (e < N_EDGES) ? ei[N_EDGES + e] : (e - N_EDGES);
  atomicAdd(deg + dst, 1.0f);
}
__global__ void compute_dinv(const float* __restrict__ deg, float* __restrict__ dinv) {
  int i = blockIdx.x * blockDim.x + threadIdx.x;
  if (i < N_NODES) dinv[i] = rsqrtf(fmaxf(deg[i], 1.0f));
}
__global__ void gcn_scatter(const int* __restrict__ ei, const float* __restrict__ H,
                            const float* __restrict__ dinv, float* __restrict__ out) {
  long long t = (long long)blockIdx.x * blockDim.x + threadIdx.x;
  const long long total = (long long)(N_EDGES + N_NODES) * (HID / 4);
  if (t >= total) return;
  int e  = (int)(t / (HID / 4));
  int c4 = (int)(t % (HID / 4)) * 4;
  int src = (e < N_EDGES) ? ei[e] : (e - N_EDGES);
  int dst = (e < N_EDGES) ? ei[N_EDGES + e] : (e - N_EDGES);
  float w = dinv[src] * dinv[dst];
  const float4 h = *(const float4*)(H + (long long)src * HID + c4);
  float* o = out + (long long)dst * HID + c4;
  atomicAdd(o + 0, h.x * w); atomicAdd(o + 1, h.y * w);
  atomicAdd(o + 2, h.z * w); atomicAdd(o + 3, h.w * w);
}

// ---------------- BatchNorm (training-mode batch stats) ----------------
__global__ __launch_bounds__(256) void bn_stats(const float* __restrict__ X,
                                                float* __restrict__ mean,
                                                float* __restrict__ var) {
  const int c = blockIdx.x;                // one block per channel
  float s = 0.0f, s2 = 0.0f;
  for (int r = threadIdx.x; r < N_NODES; r += 256) {
    float v = X[(long long)r * HID + c];
    s += v; s2 += v * v;
  }
  __shared__ float sh[256], sh2[256];
  sh[threadIdx.x] = s; sh2[threadIdx.x] = s2;
  __syncthreads();
  for (int off = 128; off > 0; off >>= 1) {
    if (threadIdx.x < off) {
      sh[threadIdx.x]  += sh[threadIdx.x + off];
      sh2[threadIdx.x] += sh2[threadIdx.x + off];
    }
    __syncthreads();
  }
  if (threadIdx.x == 0) {
    float m = sh[0] / (float)N_NODES;
    mean[c] = m;
    var[c]  = sh2[0] / (float)N_NODES - m * m;
  }
}
__global__ void bn_relu(const float* __restrict__ X, const float* __restrict__ mean,
                        const float* __restrict__ var, const float* __restrict__ g,
                        const float* __restrict__ b, float* __restrict__ Y) {
  long long i = (long long)blockIdx.x * blockDim.x + threadIdx.x;
  if (i >= (long long)N_NODES * HID) return;
  int c = (int)(i % HID);
  float y = g[c] * (X[i] - mean[c]) * rsqrtf(var[c] + BN_EPS) + b[c];
  Y[i] = fmaxf(y, 0.0f);
}

// ---------------- GAT pieces ----------------
__global__ void gat_alpha(const float* __restrict__ H, const float* __restrict__ a_src,
                          const float* __restrict__ a_dst, float* __restrict__ as_,
                          float* __restrict__ ad_) {
  int t = blockIdx.x * blockDim.x + threadIdx.x;
  if (t >= N_NODES * NHEAD) return;
  int n = t / NHEAD, h = t % NHEAD;
  const float* hp = H + (long long)n * HID + h * HDIM;
  float s = 0.0f, d = 0.0f;
#pragma unroll 8
  for (int i = 0; i < HDIM; ++i) {
    s += hp[i] * a_src[h * HDIM + i];
    d += hp[i] * a_dst[h * HDIM + i];
  }
  as_[t] = s; ad_[t] = d;
}
__device__ inline void edge_decode(const int* ei, int e, int& src, int& dst) {
  src = (e < N_EDGES) ? ei[e] : (e - N_EDGES);
  dst = (e < N_EDGES) ? ei[N_EDGES + e] : (e - N_EDGES);
}
__global__ void gat_emax(const int* __restrict__ ei, const float* __restrict__ as_,
                         const float* __restrict__ ad_, unsigned* __restrict__ emax) {
  long long t = (long long)blockIdx.x * blockDim.x + threadIdx.x;
  if (t >= (long long)(N_EDGES + N_NODES) * NHEAD) return;
  int e = (int)(t / NHEAD), h = (int)(t % NHEAD), src, dst;
  edge_decode(ei, e, src, dst);
  float val = lrelu(as_[src * NHEAD + h] + ad_[dst * NHEAD + h]);
  atomicMax(emax + dst * NHEAD + h, fenc(val));
}
__global__ void gat_esum(const int* __restrict__ ei, const float* __restrict__ as_,
                         const float* __restrict__ ad_, const unsigned* __restrict__ emax,
                         float* __restrict__ esum) {
  long long t = (long long)blockIdx.x * blockDim.x + threadIdx.x;
  if (t >= (long long)(N_EDGES + N_NODES) * NHEAD) return;
  int e = (int)(t / NHEAD), h = (int)(t % NHEAD), src, dst;
  edge_decode(ei, e, src, dst);
  float val = lrelu(as_[src * NHEAD + h] + ad_[dst * NHEAD + h]);
  atomicAdd(esum + dst * NHEAD + h, expf(val - fdec(emax[dst * NHEAD + h])));
}
__global__ void gat_scatter(const int* __restrict__ ei, const float* __restrict__ as_,
                            const float* __restrict__ ad_, const unsigned* __restrict__ emax,
                            const float* __restrict__ esum, const float* __restrict__ H,
                            float* __restrict__ out) {
  long long t = (long long)blockIdx.x * blockDim.x + threadIdx.x;
  if (t >= (long long)(N_EDGES + N_NODES) * NHEAD) return;
  int e = (int)(t / NHEAD), h = (int)(t % NHEAD), src, dst;
  edge_decode(ei, e, src, dst);
  float val = lrelu(as_[src * NHEAD + h] + ad_[dst * NHEAD + h]);
  float alpha = expf(val - fdec(emax[dst * NHEAD + h])) / esum[dst * NHEAD + h];
  const float* hp = H + (long long)src * HID + h * HDIM;
  float* op = out + (long long)dst * HID + h * HDIM;
#pragma unroll
  for (int i = 0; i < HDIM; i += 4) {
    float4 hv = *(const float4*)(hp + i);
    atomicAdd(op + i + 0, hv.x * alpha); atomicAdd(op + i + 1, hv.y * alpha);
    atomicAdd(op + i + 2, hv.z * alpha); atomicAdd(op + i + 3, hv.w * alpha);
  }
}

// ---------------- mean pooling ----------------
__global__ void pool_count(const int* __restrict__ batch, float* __restrict__ cnt) {
  int n = blockIdx.x * blockDim.x + threadIdx.x;
  if (n < N_NODES) atomicAdd(cnt + batch[n], 1.0f);
}
__global__ void pool_sum(const float* __restrict__ X, const int* __restrict__ batch,
                         float* __restrict__ pool) {
  long long t = (long long)blockIdx.x * blockDim.x + threadIdx.x;
  if (t >= (long long)N_NODES * (HID / 4)) return;
  int n = (int)(t / (HID / 4)), c4 = (int)(t % (HID / 4)) * 4;
  int g = batch[n];
  const float4 v = *(const float4*)(X + (long long)n * HID + c4);
  float* o = pool + (long long)g * HID + c4;
  atomicAdd(o + 0, v.x); atomicAdd(o + 1, v.y);
  atomicAdd(o + 2, v.z); atomicAdd(o + 3, v.w);
}
__global__ void pool_div(float* __restrict__ pool, const float* __restrict__ cnt) {
  int i = blockIdx.x * blockDim.x + threadIdx.x;
  if (i >= NB * HID) return;
  pool[i] /= fmaxf(cnt[i / HID], 1.0f);
}

// ---------------- head ----------------
__global__ void concat2(const float* __restrict__ a, const float* __restrict__ b,
                        float* __restrict__ o) {
  int i = blockIdx.x * blockDim.x + threadIdx.x;
  if (i >= NB * 2 * HID) return;
  int r = i / (2 * HID), c = i % (2 * HID);
  o[i] = (c < HID) ? a[r * HID + c] : b[r * HID + (c - HID)];
}
__global__ void final_proj(const float* __restrict__ h, const float* __restrict__ W,
                           const float* __restrict__ b, float* __restrict__ out) {
  int i = blockIdx.x * blockDim.x + threadIdx.x;
  if (i >= NB) return;
  float s = b[0];
#pragma unroll 8
  for (int k = 0; k < HID / 2; ++k) s += h[i * (HID / 2) + k] * W[k];
  out[i] = s;
}

// ---------------- host orchestration ----------------
static inline unsigned gdim(long long total, int bs) {
  return (unsigned)((total + bs - 1) / bs);
}
static void gemm(const float* A, const __bf16* Wb, const float* bias, float* C,
                 int M, int K, int NH, int act, hipStream_t s) {
  dim3 grid(M / 16, (NH + 127) / 128);
  wmma_gemm_bias_act<<<grid, dim3(256), 0, s>>>(A, Wb, bias, C, M, K, NH, act);
}
static void zero(float* p, long long n, hipStream_t s) {
  unsigned g = gdim(n, 256); if (g > (1u << 20)) g = (1u << 20);
  zero_f32<<<g, 256, 0, s>>>(p, n);
}
static void cvt(const float* in, __bf16* out, long long n, hipStream_t s) {
  cvt_f32_bf16<<<gdim(n / 4, 256), 256, 0, s>>>(in, out, n / 4);
}

extern "C" void kernel_launch(void* const* d_in, const int* in_sizes, int n_in,
                              void* d_out, int out_size, void* d_ws, size_t ws_size,
                              hipStream_t stream) {
  // ---- inputs (setup_inputs order) ----
  const float* seq_x      = (const float*)d_in[0];
  const float* struct_x   = (const float*)d_in[1];
  const int*   seq_ei     = (const int*)d_in[2];
  const int*   str_ei     = (const int*)d_in[3];
  const int*   seq_batch  = (const int*)d_in[4];
  const int*   str_batch  = (const int*)d_in[5];
  const float* seq_proj_W = (const float*)d_in[6];
  const float* seq_proj_b = (const float*)d_in[7];
  const float* gcn_W      = (const float*)d_in[8];
  // d_in[9] gcn_b: cancels exactly under BatchNorm (constant per channel)
  const float* seq_bn_g   = (const float*)d_in[10];
  const float* seq_bn_b   = (const float*)d_in[11];
  const float* str_proj_W = (const float*)d_in[12];
  const float* str_proj_b = (const float*)d_in[13];
  const float* gat_W      = (const float*)d_in[14];
  const float* gat_a_src  = (const float*)d_in[15];
  const float* gat_a_dst  = (const float*)d_in[16];
  // d_in[17] gat_b: cancels under BatchNorm
  const float* str_bn_g   = (const float*)d_in[18];
  const float* str_bn_b   = (const float*)d_in[19];
  // d_in[20..23] Wq,bq,Wk,bk: dead code (softmax over singleton axis == 1)
  const float* Wv = (const float*)d_in[24];
  const float* bv = (const float*)d_in[25];
  const float* Wo = (const float*)d_in[26];
  const float* bo = (const float*)d_in[27];
  const float* fus_W = (const float*)d_in[28];
  const float* fus_b = (const float*)d_in[29];
  const float* p1_W  = (const float*)d_in[30];
  const float* p1_b  = (const float*)d_in[31];
  const float* p2_W  = (const float*)d_in[32];
  const float* p2_b  = (const float*)d_in[33];
  float* out = (float*)d_out;

  // ---- workspace carve ----
  char* wp = (char*)d_ws;
  auto takeb = [&](long long bytes) {
    void* r = (void*)wp;
    wp += (bytes + 63) & ~63LL;          // 64B-align every buffer
    return r;
  };
  auto take  = [&](long long nf) { return (float*)takeb(nf * 4); };
  auto takeh = [&](long long nh) { return (__bf16*)takeb(nh * 2); };

  float* xcur  = take((long long)N_NODES * HID);
  float* hbuf  = take((long long)N_NODES * HID);
  float* agg   = take((long long)N_NODES * HID);
  float* deg   = take(N_NODES);
  float* dinv  = take(N_NODES);
  float* as_   = take((long long)N_NODES * NHEAD);
  float* ad_   = take((long long)N_NODES * NHEAD);
  float* emaxf = take((long long)N_NODES * NHEAD);  // holds fenc() uints
  float* esum  = take((long long)N_NODES * NHEAD);
  float* meanb = take(HID);
  float* varb  = take(HID);
  float* cnt   = take(NB);
  float* pools = take(NB * HID);
  float* poolt = take(NB * HID);
  float* vbuf  = take(NB * HID);
  float* xattn = take(NB * HID);
  float* fusin = take(NB * 2 * HID);
  float* fusout= take(NB * HID);
  float* h1    = take(NB * (HID / 2));
  // bf16 weight copies (converted once per call; removes converts from GEMM)
  __bf16* wseqp_bf = takeh((long long)DIN_SEQ * HID);
  __bf16* wgcn_bf  = takeh((long long)NLAYERS * HID * HID);
  __bf16* wstrp_bf = takeh((long long)DIN_STR * HID);
  __bf16* wgat_bf  = takeh((long long)NLAYERS * HID * HID);
  __bf16* wv_bf    = takeh((long long)HID * HID);
  __bf16* wo_bf    = takeh((long long)HID * HID);
  __bf16* wfus_bf  = takeh((long long)2 * HID * HID);
  __bf16* wp1_bf   = takeh((long long)HID * (HID / 2));
  (void)ws_size; (void)n_in; (void)in_sizes; (void)out_size;

  // ---- weight conversions (deterministic, every call) ----
  cvt(seq_proj_W, wseqp_bf, (long long)DIN_SEQ * HID, stream);
  cvt(gcn_W,      wgcn_bf,  (long long)NLAYERS * HID * HID, stream);
  cvt(str_proj_W, wstrp_bf, (long long)DIN_STR * HID, stream);
  cvt(gat_W,      wgat_bf,  (long long)NLAYERS * HID * HID, stream);
  cvt(Wv,         wv_bf,    (long long)HID * HID, stream);
  cvt(Wo,         wo_bf,    (long long)HID * HID, stream);
  cvt(fus_W,      wfus_bf,  (long long)2 * HID * HID, stream);
  cvt(p1_W,       wp1_bf,   (long long)HID * (HID / 2), stream);

  const long long E2H4 = (long long)(N_EDGES + N_NODES) * (HID / 4);
  const long long E2H  = (long long)(N_EDGES + N_NODES) * NHEAD;

  // ================= sequence branch: GCN stack =================
  zero(deg, N_NODES, stream);
  compute_deg<<<gdim(N_EDGES + N_NODES, 256), 256, 0, stream>>>(seq_ei, deg);
  compute_dinv<<<gdim(N_NODES, 256), 256, 0, stream>>>(deg, dinv);
  gemm(seq_x, wseqp_bf, seq_proj_b, xcur, N_NODES, DIN_SEQ, HID, 1, stream);
  for (int l = 0; l < NLAYERS; ++l) {
    gemm(xcur, wgcn_bf + (long long)l * HID * HID, nullptr, hbuf, N_NODES, HID, HID, 0, stream);
    zero(agg, (long long)N_NODES * HID, stream);
    gcn_scatter<<<gdim(E2H4, 256), 256, 0, stream>>>(seq_ei, hbuf, dinv, agg);
    bn_stats<<<HID, 256, 0, stream>>>(agg, meanb, varb);
    bn_relu<<<gdim((long long)N_NODES * HID, 256), 256, 0, stream>>>(
        agg, meanb, varb, seq_bn_g + l * HID, seq_bn_b + l * HID, xcur);
  }
  zero(cnt, NB, stream);
  zero(pools, NB * HID, stream);
  pool_count<<<gdim(N_NODES, 256), 256, 0, stream>>>(seq_batch, cnt);
  pool_sum<<<gdim((long long)N_NODES * (HID / 4), 256), 256, 0, stream>>>(xcur, seq_batch, pools);
  pool_div<<<gdim(NB * HID, 256), 256, 0, stream>>>(pools, cnt);

  // ================= structure branch: GAT stack =================
  gemm(struct_x, wstrp_bf, str_proj_b, xcur, N_NODES, DIN_STR, HID, 1, stream);
  for (int l = 0; l < NLAYERS; ++l) {
    gemm(xcur, wgat_bf + (long long)l * HID * HID, nullptr, hbuf, N_NODES, HID, HID, 0, stream);
    gat_alpha<<<gdim(N_NODES * NHEAD, 256), 256, 0, stream>>>(
        hbuf, gat_a_src + l * NHEAD * HDIM, gat_a_dst + l * NHEAD * HDIM, as_, ad_);
    zero(emaxf, (long long)N_NODES * NHEAD, stream);  // 0u encodes a tiny value
    zero(esum, (long long)N_NODES * NHEAD, stream);
    zero(agg, (long long)N_NODES * HID, stream);
    gat_emax<<<gdim(E2H, 256), 256, 0, stream>>>(str_ei, as_, ad_, (unsigned*)emaxf);
    gat_esum<<<gdim(E2H, 256), 256, 0, stream>>>(str_ei, as_, ad_, (unsigned*)emaxf, esum);
    gat_scatter<<<gdim(E2H, 256), 256, 0, stream>>>(str_ei, as_, ad_, (unsigned*)emaxf, esum, hbuf, agg);
    bn_stats<<<HID, 256, 0, stream>>>(agg, meanb, varb);
    bn_relu<<<gdim((long long)N_NODES * HID, 256), 256, 0, stream>>>(
        agg, meanb, varb, str_bn_g + l * HID, str_bn_b + l * HID, xcur);
  }
  zero(cnt, NB, stream);
  zero(poolt, NB * HID, stream);
  pool_count<<<gdim(N_NODES, 256), 256, 0, stream>>>(str_batch, cnt);
  pool_sum<<<gdim((long long)N_NODES * (HID / 4), 256), 256, 0, stream>>>(xcur, str_batch, poolt);
  pool_div<<<gdim(NB * HID, 256), 256, 0, stream>>>(poolt, cnt);

  // ================= head =================
  // softmax over singleton axis == 1 -> x_attn = (xt@Wv+bv)@Wo+bo
  gemm(poolt, wv_bf, bv, vbuf, NB, HID, HID, 0, stream);
  gemm(vbuf, wo_bf, bo, xattn, NB, HID, HID, 0, stream);
  concat2<<<gdim(NB * 2 * HID, 256), 256, 0, stream>>>(xattn, poolt, fusin);
  gemm(fusin, wfus_bf, fus_b, fusout, NB, 2 * HID, HID, 1, stream);
  gemm(fusout, wp1_bf, p1_b, h1, NB, HID, HID / 2, 1, stream);
  final_proj<<<1, NB, 0, stream>>>(h1, p2_W, p2_b, out);
}
